// CrossAttention_26749056319609
// MI455X (gfx1250) — matrix operand order repaired and checked by reading, not compiled, not run on hardware
//
#include <hip/hip_runtime.h>
#include <hip/hip_bf16.h>
#include <math.h>

// CrossAttention for MI455X (gfx1250, wave32, WMMA).
// B=8, C=Cc=256, H=W=64 -> N=M=4096, D=32.

typedef __attribute__((ext_vector_type(16))) _Float16 v16h;
typedef __attribute__((ext_vector_type(8)))  _Float16 v8h;
typedef __attribute__((ext_vector_type(2)))  __fp16   h2raw;  // cvt_pkrtz result type
typedef __attribute__((ext_vector_type(8)))  float    v8f;

#define NB 4096   // N = H*W
#define MB 4096   // M = Hc*Wc
#define CB 256    // C (= Cc)
#define DB 32     // D = C/8

__device__ __forceinline__ v16h mk16(v8h a, v8h b) {
  v16h r;
#pragma unroll
  for (int i = 0; i < 8; ++i) { r[i] = a[i]; r[i + 8] = b[i]; }
  return r;
}
__device__ __forceinline__ v8h ld8h(const _Float16* p) { return *(const v8h*)p; }
__device__ __forceinline__ v16h frag32(const _Float16* p) {  // 16 contiguous f16
  return mk16(ld8h(p), ld8h(p + 8));
}
// pack two f32 -> f16x2 (v_cvt_pk_rtz_f16_f32) and store as one b32
__device__ __forceinline__ void st_pk(_Float16* p, float a, float b) {
  *(h2raw*)p = __builtin_amdgcn_cvt_pkrtz(a, b);
}

// ---------------------------------------------------------------- cvt weights
__global__ void ca_cvt(const float* __restrict__ s, _Float16* __restrict__ d, int n) {
  int i = blockIdx.x * 256 + threadIdx.x;
  if (i < n) d[i] = (_Float16)s[i];
}

// ---------------------------------------------------------------- projection
// out^T[d][n] = sum_c W[d][c] * in[b][c][n] + bias[d]
// Workgroup: 4 waves, n-block 64 (4 subtiles), wave w owns d-tile blockIdx.y*64+w*16
// and computes all 4 n-subtiles (16 WMMA / K-step / WG vs 32 packed b32 DS stores).
__global__ __launch_bounds__(128) void ca_proj(
    const float* __restrict__ in, const _Float16* __restrict__ Wh,
    const float* __restrict__ bias, _Float16* __restrict__ outh,
    int Cin, int Dout, int transposed) {
  __shared__ _Float16 inT[64][48];   // [n][c], row stride 96B (16B aligned)

  const int b  = blockIdx.z;
  const int n0 = blockIdx.x * 64;
  const int t  = threadIdx.x;
  const int w  = t >> 5;
  const int l  = t & 31;
  const int lm = l & 15, lh = l >> 4;
  const int dt = blockIdx.y * 64 + w * 16;
  const bool active = dt < Dout;

  v8f acc[4] = {{}, {}, {}, {}};
  const float* src_base = in + (size_t)b * Cin * NB + n0;

  for (int c0 = 0; c0 < Cin; c0 += 32) {
    {  // stage 32c x 64n -> LDS f16 [n][c], packed channel pairs (b32 stores)
      const int cp  = t >> 3;          // 0..15 channel pair
      const int nb8 = (t & 7) * 8;     // 0,8,...,56
      const float* se = src_base + (size_t)(c0 + 2 * cp) * NB + nb8;
      const float* so = se + NB;
      float4 e0 = *(const float4*)(se);
      float4 e1 = *(const float4*)(se + 4);
      float4 o0 = *(const float4*)(so);
      float4 o1 = *(const float4*)(so + 4);
      st_pk(&inT[nb8 + 0][2 * cp], e0.x, o0.x);
      st_pk(&inT[nb8 + 1][2 * cp], e0.y, o0.y);
      st_pk(&inT[nb8 + 2][2 * cp], e0.z, o0.z);
      st_pk(&inT[nb8 + 3][2 * cp], e0.w, o0.w);
      st_pk(&inT[nb8 + 4][2 * cp], e1.x, o1.x);
      st_pk(&inT[nb8 + 5][2 * cp], e1.y, o1.y);
      st_pk(&inT[nb8 + 6][2 * cp], e1.z, o1.z);
      st_pk(&inT[nb8 + 7][2 * cp], e1.w, o1.w);
    }
    __syncthreads();
    if (active) {
      const _Float16* wp = Wh + (size_t)(dt + lm) * Cin + c0 + lh * 8;
      v16h Af = mk16(ld8h(wp), ld8h(wp + 16));
#pragma unroll
      for (int ns = 0; ns < 4; ++ns) {
        v16h Bf = mk16(ld8h(&inT[ns * 16 + lm][lh * 8]), ld8h(&inT[ns * 16 + lm][16 + lh * 8]));
        acc[ns] = __builtin_amdgcn_wmma_f32_16x16x32_f16(false, Af, false, Bf,
                                                         (short)0, acc[ns], false, false);
      }
    }
    __syncthreads();
  }
  if (active) {
#pragma unroll
    for (int ns = 0; ns < 4; ++ns)
#pragma unroll
      for (int r = 0; r < 8; ++r) {
        const int d = dt + r + 8 * lh;
        const int n = n0 + ns * 16 + lm;
        float v = acc[ns][r] + bias[d];
        if (transposed) outh[((size_t)b * Dout + d) * NB + n] = (_Float16)v;
        else            outh[((size_t)b * NB + n) * Dout + d] = (_Float16)v;
      }
  }
}

// ---------------------------------------------------------------- attention
// Waves: (w&1)=n-subtile (16 rows), (w>>1)=e-half (128 cols). m-chunk = 64.
// Per chunk/wave: 4 S-WMMA + 2 rowsum-WMMA + 16 PV-WMMA, 32 max-shuffles,
// packed b32 P-stores, rescale skipped when the running max is unchanged.
__global__ __launch_bounds__(128) void ca_attn(
    const _Float16* __restrict__ qh,   // [B][N][32]
    const _Float16* __restrict__ kh,   // [B][M][32]
    const _Float16* __restrict__ vth,  // [B][256][M]  (v transposed)
    const float* __restrict__ x, const float* __restrict__ gamma,
    float* __restrict__ out) {
  __shared__ _Float16 Pbuf[4][16][80];   // [n][k0..63], row stride 160B
  __shared__ float    Obuf[4][128][20];  // epilogue transpose

  const int b  = blockIdx.z;
  const int t  = threadIdx.x;
  const int w  = t >> 5;
  const int l  = t & 31;
  const int lm = l & 15, lh = l >> 4;
  const int n0 = blockIdx.x * 32 + (w & 1) * 16;
  const int e0 = (w >> 1) * 128;

  // q A-frag (once): row n = n0+lm, K contiguous d = lh*16 .. lh*16+15
  const _Float16* qp = qh + ((size_t)b * NB + n0 + lm) * DB + lh * 16;
  const v16h qA = frag32(qp);

  const _Float16* kb = kh + (size_t)b * MB * DB;
  const _Float16* vb = vth + (size_t)b * CB * MB;

  v16h ones;
#pragma unroll
  for (int i = 0; i < 16; ++i) ones[i] = (_Float16)1.0f;

  v8f acc[8] = {{}, {}, {}, {}, {}, {}, {}, {}};
  v8f lsum = {};
  float mr[8];
#pragma unroll
  for (int r = 0; r < 8; ++r) mr[r] = -1e30f;

  for (int m0 = 0; m0 < MB; m0 += 64) {
    const _Float16* kp = kb + (size_t)m0 * DB;
    // Column mapping: tile0 -> even m (m0+2*lm), tile1 -> odd; same for 2nd half.
    v8f z = {};
    v16h kB0 = frag32(kp + (size_t)(2 * lm) * DB + lh * 16);
    v16h kB1 = frag32(kp + (size_t)(2 * lm + 1) * DB + lh * 16);
    v8f S0 = __builtin_amdgcn_wmma_f32_16x16x32_f16(false, qA, false, kB0, (short)0, z, false, false);
    v8f S1 = __builtin_amdgcn_wmma_f32_16x16x32_f16(false, qA, false, kB1, (short)0, z, false, false);
    v16h kB2 = frag32(kp + (size_t)(32 + 2 * lm) * DB + lh * 16);
    v16h kB3 = frag32(kp + (size_t)(33 + 2 * lm) * DB + lh * 16);
    v8f S2 = __builtin_amdgcn_wmma_f32_16x16x32_f16(false, qA, false, kB2, (short)0, z, false, false);
    v8f S3 = __builtin_amdgcn_wmma_f32_16x16x32_f16(false, qA, false, kB3, (short)0, z, false, false);

    if (m0 + 64 < MB)  // next chunk's k rows (streaming operand)
      __builtin_prefetch(kp + (size_t)(64 + 2 * l) * DB, 0, 1);

    // running row-max (one 4-step butterfly per row over 4 tiles)
    float tm[8];
    bool upd = false;
#pragma unroll
    for (int r = 0; r < 8; ++r) {
      float t4 = fmaxf(fmaxf(S0[r], S1[r]), fmaxf(S2[r], S3[r]));
#pragma unroll
      for (int off = 8; off >= 1; off >>= 1) t4 = fmaxf(t4, __shfl_xor(t4, off, 32));
      tm[r] = t4;
      upd = upd || (t4 > mr[r]);
    }
    if (__any((int)upd)) {  // wave-uniform: rescale only when max moved
#pragma unroll
      for (int r = 0; r < 8; ++r) {
        float mn = fmaxf(mr[r], tm[r]);
        float corr = __expf(mr[r] - mn);
        mr[r] = mn;
        lsum[r] *= corr;
#pragma unroll
        for (int tt = 0; tt < 8; ++tt) acc[tt][r] *= corr;
      }
    }
    // P = exp(S - mr), packed pairs (adjacent K) -> b32 LDS stores
#pragma unroll
    for (int r = 0; r < 8; ++r) {
      float p0 = __expf(S0[r] - mr[r]);
      float p1 = __expf(S1[r] - mr[r]);
      float p2 = __expf(S2[r] - mr[r]);
      float p3 = __expf(S3[r] - mr[r]);
      st_pk(&Pbuf[w][r + 8 * lh][2 * lm],      p0, p1);
      st_pk(&Pbuf[w][r + 8 * lh][32 + 2 * lm], p2, p3);
    }
    asm volatile("s_wait_dscnt 0" ::: "memory");  // same-wave DS is in-order; be explicit
    const _Float16* pp = &Pbuf[w][lm][0];
    v16h pA0 = mk16(*(const v8h*)(pp + lh * 16),      *(const v8h*)(pp + 8 + lh * 16));
    v16h pA1 = mk16(*(const v8h*)(pp + 32 + lh * 16), *(const v8h*)(pp + 40 + lh * 16));
    // rowsum(P) via WMMA with ones-B: lands broadcast in C-layout rows
    lsum = __builtin_amdgcn_wmma_f32_16x16x32_f16(false, pA0, false, ones, (short)0, lsum, false, false);
    lsum = __builtin_amdgcn_wmma_f32_16x16x32_f16(false, pA1, false, ones, (short)0, lsum, false, false);
    // PV: col e = e0 + tt*16 + lm, K chunks m = m0 + lh*16 (+32)
#pragma unroll
    for (int tt = 0; tt < 8; ++tt) {
      const _Float16* vp = vb + (size_t)(e0 + tt * 16 + lm) * MB + m0 + lh * 16;
      acc[tt] = __builtin_amdgcn_wmma_f32_16x16x32_f16(false, pA0, false, frag32(vp),
                                                       (short)0, acc[tt], false, false);
      acc[tt] = __builtin_amdgcn_wmma_f32_16x16x32_f16(false, pA1, false, frag32(vp + 32),
                                                       (short)0, acc[tt], false, false);
    }
  }

  // epilogue: out[b][e][n] = gamma * acc/lsum + x ; LDS transpose for coalescing
  const float g = gamma[0];
  float rinv[8];
#pragma unroll
  for (int r = 0; r < 8; ++r) rinv[r] = 1.0f / lsum[r];
#pragma unroll
  for (int tt = 0; tt < 8; ++tt)
#pragma unroll
    for (int r = 0; r < 8; ++r)
      Obuf[w][tt * 16 + lm][r + 8 * lh] = acc[tt][r] * rinv[r];
  asm volatile("s_wait_dscnt 0" ::: "memory");
#pragma unroll
  for (int j = 0; j < 4; ++j) {
    const int erow = l + 32 * j;             // 0..127
    const int e = e0 + erow;
    const float* xr = x   + ((size_t)b * CB + e) * NB + n0;
    float*      op = out + ((size_t)b * CB + e) * NB + n0;
#pragma unroll
    for (int q4 = 0; q4 < 4; ++q4) {
      float4 xv = *(const float4*)(xr + q4 * 4);
      float4 ov;
      ov.x = g * Obuf[w][erow][q4 * 4 + 0] + xv.x;
      ov.y = g * Obuf[w][erow][q4 * 4 + 1] + xv.y;
      ov.z = g * Obuf[w][erow][q4 * 4 + 2] + xv.z;
      ov.w = g * Obuf[w][erow][q4 * 4 + 3] + xv.w;
      *(float4*)(op + q4 * 4) = ov;
    }
  }
}

// ---------------------------------------------------------------- launch
extern "C" void kernel_launch(void* const* d_in, const int* in_sizes, int n_in,
                              void* d_out, int out_size, void* d_ws, size_t ws_size,
                              hipStream_t stream) {
  const float* x     = (const float*)d_in[0];
  const float* cond  = (const float*)d_in[1];
  const float* Wq    = (const float*)d_in[2];
  const float* bq    = (const float*)d_in[3];
  const float* Wk    = (const float*)d_in[4];
  const float* bk    = (const float*)d_in[5];
  const float* Wv    = (const float*)d_in[6];
  const float* bv    = (const float*)d_in[7];
  const float* gamma = (const float*)d_in[8];
  float* out = (float*)d_out;

  char* ws = (char*)d_ws;
  _Float16* Wqh = (_Float16*)ws; ws += (size_t)DB * CB * 2;
  _Float16* Wkh = (_Float16*)ws; ws += (size_t)DB * CB * 2;
  _Float16* Wvh = (_Float16*)ws; ws += (size_t)CB * CB * 2;
  _Float16* qhp = (_Float16*)ws; ws += (size_t)8 * NB * DB * 2;
  _Float16* khp = (_Float16*)ws; ws += (size_t)8 * MB * DB * 2;
  _Float16* vtp = (_Float16*)ws;  // 16 MB

  ca_cvt<<<(DB * CB + 255) / 256, 256, 0, stream>>>(Wq, Wqh, DB * CB);
  ca_cvt<<<(DB * CB + 255) / 256, 256, 0, stream>>>(Wk, Wkh, DB * CB);
  ca_cvt<<<(CB * CB + 255) / 256, 256, 0, stream>>>(Wv, Wvh, CB * CB);

  dim3 gqk(NB / 64, 1, 8);          // Dout=32 -> 1 y-tile (waves 2,3 stage only)
  ca_proj<<<gqk, 128, 0, stream>>>(x,    Wqh, bq, qhp, CB, DB, 0);
  ca_proj<<<gqk, 128, 0, stream>>>(cond, Wkh, bk, khp, CB, DB, 0);
  dim3 gv(MB / 64, CB / 64, 8);     // Dout=256 -> 4 y-tiles
  ca_proj<<<gv, 128, 0, stream>>>(cond, Wvh, bv, vtp, CB, CB, 1);

  dim3 ga(NB / 32, 1, 8);
  ca_attn<<<ga, 128, 0, stream>>>(qhp, khp, vtp, x, gamma, out);
}
